// CorrespondenceExtractor_2173253452295
// MI455X (gfx1250) — matrix operand order repaired and checked by reading, not compiled, not run on hardware
//
#include <hip/hip_runtime.h>
#include <hip/hip_bf16.h>

// ---------------------------------------------------------------------------
// CorrespondenceExtractor on MI455X (gfx1250)
//   * feature-kNN (k=2) via v_wmma_f32_16x16x32_f16, fused top-2 reduction
//   * steady loop (per 2 col-tiles): 4 loads + 4 WMMA + 2x(max-tree+cmp+branch)
//   * Lowe ratio weights, exact top-2048 rank selection, fused gather
// ---------------------------------------------------------------------------

#define NQ        16384
#define MS        16384
#define C_DIM     64
#define NUM_CORR  2048
#define EPS_W     1e-8f

// d_out layout (floats), reference return order, 4096 = 2*NUM_CORR rows:
#define OUT_SP 0          // src_corr_points [4096,3]
#define OUT_TP 12288      // tgt_corr_points [4096,3]
#define OUT_SF 24576      // src_corr_feats  [4096,64]
#define OUT_TF 286720     // tgt_corr_feats  [4096,64]
#define OUT_W  548864     // corr_weights    [4096]

typedef __attribute__((ext_vector_type(16))) _Float16 v16h;
typedef __attribute__((ext_vector_type(8)))  _Float16 v8h;
typedef __attribute__((ext_vector_type(8)))  float    v8f;

// ---------------------------------------------------------------------------
__global__ void cvt_f32_to_f16(const float* __restrict__ in,
                               _Float16* __restrict__ out, int n) {
  int i = blockIdx.x * blockDim.x + threadIdx.x;
  if (i < n) out[i] = (_Float16)in[i];
}

// ---------------------------------------------------------------------------
// kNN top-2 by dot product.  One wave owns a 16-row tile of Q, sweeps all
// 16-column tiles of S.  K=64 -> two v_wmma_f32_16x16x32_f16 per tile.
//
// A (16x32 f16): lane m / m+16 both hold row m;
//   lane m: K = s*32 + {0..15};  lane m+16: K = s*32 + {16..31}
// B (32x16 f16): lane n holds col n K={0..15}; lane n+16 col n K={16..31}
// C (16x16 f32): reg r, lanes 0-15 -> row r, col lane; lanes 16-31 -> row r+8
// ---------------------------------------------------------------------------
__device__ __forceinline__ bool pair_better(float av, int ai, float bv, int bi) {
  return (av > bv) || (av == bv && ai < bi);
}

// top-2 update for one 16x16 C tile; quick-check shares no subexpression with
// the guarded update so nothing gets hoisted into the steady-state loop.
#define TOP2_UPDATE(ACC, COLBASE)                                             \
  {                                                                           \
    float m_ = fmaxf(fmaxf(fmaxf((ACC)[0], (ACC)[1]), fmaxf((ACC)[2], (ACC)[3])), \
                     fmaxf(fmaxf((ACC)[4], (ACC)[5]), fmaxf((ACC)[6], (ACC)[7]))); \
    if (m_ > vmin2) {                 /* rare after warm-up */                \
      const int col_ = (COLBASE) + l15;                                       \
      _Pragma("unroll")                                                       \
      for (int r = 0; r < 8; ++r) {                                           \
        float v_ = (ACC)[r];                                                  \
        bool g1_ = v_ > v1[r];                                                \
        bool g0_ = v_ > v0[r];                                                \
        int  ic_ = g1_ ? col_ : i1[r];                                        \
        i1[r] = g0_ ? i0[r] : ic_;                                            \
        i0[r] = g0_ ? col_  : i0[r];                                          \
        v1[r] = fmaxf(v1[r], fminf(v_, v0[r]));                               \
        v0[r] = fmaxf(v0[r], v_);                                             \
      }                                                                       \
      vmin2 = fminf(fminf(fminf(v1[0], v1[1]), fminf(v1[2], v1[3])),          \
                    fminf(fminf(v1[4], v1[5]), fminf(v1[6], v1[7])));         \
    }                                                                         \
  }

__global__ __launch_bounds__(256)
void knn_top2_wmma(const _Float16* __restrict__ qf,   // [NQ, 64] f16
                   const _Float16* __restrict__ sf,   // [MS, 64] f16
                   float* __restrict__ outW,          // [NQ] ratio-test weight
                   int*   __restrict__ outNN) {       // [NQ] nearest index
  const int lane    = threadIdx.x & 31;
  const int wave    = threadIdx.x >> 5;
  const int rowBase = (blockIdx.x * 8 + wave) * 16;
  const int l15     = lane & 15;
  const int half    = lane >> 4;
  const int NT      = MS / 16;

  // ---- A fragments, resident for the whole sweep ----
  const _Float16* qrow = qf + (size_t)(rowBase + l15) * C_DIM;
  v8h a0lo = *(const v8h*)(qrow +  0 + half * 16);
  v8h a0hi = *(const v8h*)(qrow +  8 + half * 16);
  v8h a1lo = *(const v8h*)(qrow + 32 + half * 16);
  v8h a1hi = *(const v8h*)(qrow + 40 + half * 16);
  v16h a0 = __builtin_shufflevector(a0lo, a0hi, 0,1,2,3,4,5,6,7,8,9,10,11,12,13,14,15);
  v16h a1 = __builtin_shufflevector(a1lo, a1hi, 0,1,2,3,4,5,6,7,8,9,10,11,12,13,14,15);

  // ---- per-lane running top-2 (value, col index) for 8 C registers ----
  float v0[8], v1[8];
  int   i0[8], i1[8];
#pragma unroll
  for (int r = 0; r < 8; ++r) { v0[r] = -1e30f; v1[r] = -1e30f; i0[r] = 0; i1[r] = 0; }
  float vmin2 = -1e30f;             // min over v1[0..7] (lane's weakest 2nd-best)

  // ---- software pipeline: two buffer sets, reloaded in place (no copies) ---
  // NOTE: loads run up to 2 tiles (4 KB) past the end of sf; workspace layout
  // pads for this and the over-read values are never consumed.
  const _Float16* srow0 = sf + (size_t)l15 * C_DIM + half * 16;
  const size_t tileStride = (size_t)16 * C_DIM;
  v16h b0a = *(const v16h*)(srow0);
  v16h b1a = *(const v16h*)(srow0 + 32);
  v16h b0b = *(const v16h*)(srow0 + tileStride);
  v16h b1b = *(const v16h*)(srow0 + tileStride + 32);

  for (int ct = 0; ct < NT; ct += 2) {
    const _Float16* pA = srow0 + (size_t)(ct + 2) * tileStride;
    const _Float16* pB = srow0 + (size_t)(ct + 3) * tileStride;

    v8f accA = {};
    accA = __builtin_amdgcn_wmma_f32_16x16x32_f16(false, a0, false, b0a,
                                                  (short)0, accA, false, false);
    accA = __builtin_amdgcn_wmma_f32_16x16x32_f16(false, a1, false, b1a,
                                                  (short)0, accA, false, false);
    b0a = *(const v16h*)(pA);
    b1a = *(const v16h*)(pA + 32);

    v8f accB = {};
    accB = __builtin_amdgcn_wmma_f32_16x16x32_f16(false, a0, false, b0b,
                                                  (short)0, accB, false, false);
    accB = __builtin_amdgcn_wmma_f32_16x16x32_f16(false, a1, false, b1b,
                                                  (short)0, accB, false, false);
    b0b = *(const v16h*)(pB);
    b1b = *(const v16h*)(pB + 32);

    __builtin_prefetch(srow0 + (size_t)(ct + 4) * tileStride, 0, 3);

    TOP2_UPDATE(accA, ct * 16);
    TOP2_UPDATE(accB, (ct + 1) * 16);
  }

  // ---- merge top-2 across the 16 lanes sharing each row (exact tie-break) --
#pragma unroll
  for (int r = 0; r < 8; ++r) {
#pragma unroll
    for (int off = 1; off < 16; off <<= 1) {
      float ov0 = __shfl_xor(v0[r], off, 32);
      int   oi0 = __shfl_xor(i0[r], off, 32);
      float ov1 = __shfl_xor(v1[r], off, 32);
      int   oi1 = __shfl_xor(i1[r], off, 32);
      if (pair_better(ov0, oi0, v0[r], i0[r])) {
        float tv = v0[r]; int ti = i0[r];
        v0[r] = ov0; i0[r] = oi0; ov0 = tv; oi0 = ti;
        tv = v1[r]; ti = i1[r];
        v1[r] = ov1; i1[r] = oi1; ov1 = tv; oi1 = ti;
      }
      if (pair_better(ov0, oi0, v1[r], i1[r])) { v1[r] = ov0; i1[r] = oi0; }
    }
  }

  // ---- Lowe ratio weight; lane 0 -> rows 0..7, lane 16 -> rows 8..15 ----
  if (l15 == 0) {
#pragma unroll
    for (int r = 0; r < 8; ++r) {
      int row    = rowBase + r + half * 8;
      float sim0 = 1.0f - v0[r];            // similarity = 1 - dot
      float sim1 = 1.0f - v1[r];
      outW[row]  = 1.0f - sim0 / (sim1 + EPS_W);
      outNN[row] = i0[r];
    }
  }
}

// ---------------------------------------------------------------------------
// Exact top-NUM_CORR selection by rank counting (stable: value desc, index
// asc tie-break == lax.top_k order), fused with the output gathers.
// ---------------------------------------------------------------------------
__global__ __launch_bounds__(256)
void select_gather(const float* __restrict__ w, const int* __restrict__ nn,
                   const float* __restrict__ srcPts, const float* __restrict__ tgtPts,
                   const float* __restrict__ srcFts, const float* __restrict__ tgtFts,
                   float* __restrict__ out, int q_is_src, int slotBase) {
  __shared__ float wtile[256];
  const int i  = blockIdx.x * 256 + threadIdx.x;
  const float wi = w[i];

  int rank = 0;
  for (int base = 0; base < NQ; base += 256) {
    wtile[threadIdx.x] = w[base + threadIdx.x];
    __syncthreads();
#pragma unroll 8
    for (int j = 0; j < 256; ++j) {
      float wj = wtile[j];
      int   jg = base + j;
      rank += (int)((wj > wi) || (wj == wi && jg < i));
    }
    __syncthreads();
  }

  if (rank < NUM_CORR) {
    const int slot = slotBase + rank;
    const int nnI  = nn[i];
    const int si   = q_is_src ? i   : nnI;   // index into src arrays
    const int ti   = q_is_src ? nnI : i;     // index into tgt arrays
#pragma unroll
    for (int d = 0; d < 3; ++d) {
      out[OUT_SP + slot * 3 + d] = srcPts[si * 3 + d];
      out[OUT_TP + slot * 3 + d] = tgtPts[ti * 3 + d];
    }
    for (int c = 0; c < C_DIM; ++c) {
      out[OUT_SF + slot * C_DIM + c] = srcFts[si * C_DIM + c];
      out[OUT_TF + slot * C_DIM + c] = tgtFts[ti * C_DIM + c];
    }
    out[OUT_W + slot] = wi;
  }
}

// ---------------------------------------------------------------------------
extern "C" void kernel_launch(void* const* d_in, const int* in_sizes, int n_in,
                              void* d_out, int out_size, void* d_ws, size_t ws_size,
                              hipStream_t stream) {
  const float* srcP = (const float*)d_in[0];   // [NQ,3]
  const float* tgtP = (const float*)d_in[1];   // [MS,3]
  const float* srcF = (const float*)d_in[2];   // [NQ,64]
  const float* tgtF = (const float*)d_in[3];   // [MS,64]
  float* out = (float*)d_out;
  char*  ws  = (char*)d_ws;

  // workspace layout (bytes); each f16 region has >=64KB of ws after it, which
  // absorbs the pipeline's 2-tile (4 KB) over-read.  ~4.3 MB total.
  _Float16* sf16 = (_Float16*)(ws + 0);                       // 2 MB
  _Float16* tf16 = (_Float16*)(ws + 2097152);                 // 2 MB
  float*    wS   = (float*)   (ws + 4194304);                 // 64 KB
  int*      nnS  = (int*)     (ws + 4259840);                 // 64 KB
  float*    wT   = (float*)   (ws + 4325376);                 // 64 KB
  int*      nnT  = (int*)     (ws + 4390912);                 // 64 KB (+4 KB pad after)

  const int nfeat = NQ * C_DIM;
  cvt_f32_to_f16<<<nfeat / 256, 256, 0, stream>>>(srcF, sf16, nfeat);
  cvt_f32_to_f16<<<nfeat / 256, 256, 0, stream>>>(tgtF, tf16, nfeat);

  // 1024 row tiles per side, 8 waves (tiles) per 256-thread block
  knn_top2_wmma<<<NQ / 16 / 8, 256, 0, stream>>>(sf16, tf16, wS, nnS);
  knn_top2_wmma<<<MS / 16 / 8, 256, 0, stream>>>(tf16, sf16, wT, nnT);

  // side 1: query = src -> slots [0, 2048); side 2: query = tgt -> [2048, 4096)
  select_gather<<<NQ / 256, 256, 0, stream>>>(wS, nnS, srcP, tgtP, srcF, tgtF,
                                              out, /*q_is_src=*/1, /*slotBase=*/0);
  select_gather<<<MS / 256, 256, 0, stream>>>(wT, nnT, srcP, tgtP, srcF, tgtF,
                                              out, /*q_is_src=*/0, /*slotBase=*/NUM_CORR);
}
